// MultiHeadAttention_25494925869654
// MI455X (gfx1250) — matrix operand order repaired and checked by reading, not compiled, not run on hardware
//
#include <hip/hip_runtime.h>
#include <hip/hip_bf16.h>

typedef __bf16 bf16;
typedef __attribute__((ext_vector_type(16))) __bf16 v16bf;
typedef __attribute__((ext_vector_type(8)))  float  v8f;
typedef __attribute__((__vector_size__(16))) int    v4i;

#define B_    32
#define HEADS 8
#define DT    256
#define DM    1024
#define KS    13
#define PADR  6
#define LPAD  (DM + 2*PADR)   // 1036
#define KCONV (KS * DT)       // 3328

#if defined(__gfx1250__) && __has_builtin(__builtin_amdgcn_global_load_async_to_lds_b128) && __has_builtin(__builtin_amdgcn_s_wait_asynccnt)
#define USE_ASYNC_LDS 1
#else
#define USE_ASYNC_LDS 0
#endif

// ---------------------------------------------------------------------------
// Generic NT bf16 WMMA GEMM:  C[M][N] = alpha * sum_k A[M][k]*B[N][k] + bias_m
// Block tile 128x128, 8 waves, wave tile 64x32 (4x2 wmma tiles).
// Double-buffered LDS; async global->LDS copies when the toolchain has them.
// Batched over blockIdx.z: z -> (h = z/nB, b = z%nB) with independent strides.
// ---------------------------------------------------------------------------
__global__ __launch_bounds__(256)
void gemm_nt_bf16(const bf16* __restrict__ A, const bf16* __restrict__ B,
                  float* __restrict__ Cf, bf16* __restrict__ Cb,
                  int M, int N, int K, int lda, int ldb, int ldc, int nB,
                  long sAh, long sAb, long sBh, long sBb, long sCh, long sCb,
                  const float* __restrict__ bias_m, float alpha) {
  int z = blockIdx.z;
  int bb = z % nB, hh = z / nB;
  A += hh * sAh + bb * sAb;
  B += hh * sBh + bb * sBb;
  long cBase = hh * sCh + bb * sCb;

  __shared__ __align__(16) bf16 As[2][128][32];
  __shared__ __align__(16) bf16 Bs[2][128][32];

  int tid  = threadIdx.x;
  int lane = tid & 31, wave = tid >> 5;
  int wm = wave >> 2, wn = wave & 3;            // 2 x 4 wave grid
  int m0 = blockIdx.y * 128, n0 = blockIdx.x * 128;
  int lr = tid >> 1;                            // 0..127 staging row
  int lk = (tid & 1) * 16;                      // 0 or 16 (bf16 elems)
  int rlo = lane & 15, sel = lane >> 4;

  const bf16* aRow = A + (long)(m0 + lr) * lda + lk;   // + k0 per step
  const bf16* bRow = B + (long)(n0 + lr) * ldb + lk;

  v8f acc[4][2];
#pragma unroll
  for (int i = 0; i < 4; ++i)
#pragma unroll
    for (int j = 0; j < 2; ++j) acc[i][j] = (v8f)0.0f;

  // stage one 128x32 A tile + 128x32 B tile into LDS buffer `buf`
  auto stage = [&](int buf, int k0) {
#if USE_ASYNC_LDS
    __attribute__((address_space(1))) v4i* ga =
        (__attribute__((address_space(1))) v4i*)(aRow + k0);
    __attribute__((address_space(1))) v4i* gb =
        (__attribute__((address_space(1))) v4i*)(bRow + k0);
    __attribute__((address_space(3))) v4i* la =
        (__attribute__((address_space(3))) v4i*)&As[buf][lr][lk];
    __attribute__((address_space(3))) v4i* lb =
        (__attribute__((address_space(3))) v4i*)&Bs[buf][lr][lk];
    __builtin_amdgcn_global_load_async_to_lds_b128(ga,     la,     0, 0);
    __builtin_amdgcn_global_load_async_to_lds_b128(ga + 1, la + 1, 0, 0);
    __builtin_amdgcn_global_load_async_to_lds_b128(gb,     lb,     0, 0);
    __builtin_amdgcn_global_load_async_to_lds_b128(gb + 1, lb + 1, 0, 0);
#else
    float4 a0 = *(const float4*)(aRow + k0);
    float4 a1 = *(const float4*)(aRow + k0 + 8);
    float4 b0 = *(const float4*)(bRow + k0);
    float4 b1 = *(const float4*)(bRow + k0 + 8);
    *(float4*)&As[buf][lr][lk]     = a0;
    *(float4*)&As[buf][lr][lk + 8] = a1;
    *(float4*)&Bs[buf][lr][lk]     = b0;
    *(float4*)&Bs[buf][lr][lk + 8] = b1;
#endif
  };

  int nk = K >> 5;
  stage(0, 0);
#if USE_ASYNC_LDS
  __builtin_amdgcn_s_wait_asynccnt(0);
#endif
  __syncthreads();

  for (int t = 0; t < nk; ++t) {
    int cur = t & 1;
    if (t + 1 < nk) stage(cur ^ 1, (t + 1) << 5);

    union Frag { v16bf v; float4 q[2]; };
    Frag af[4], bfr[2];
#pragma unroll
    for (int i = 0; i < 4; ++i) {                 // A: lane holds row, K chunks
      int row = wm * 64 + i * 16 + rlo;
      af[i].q[0] = *(const float4*)&As[cur][row][sel * 8];
      af[i].q[1] = *(const float4*)&As[cur][row][16 + sel * 8];
    }
#pragma unroll
    for (int j = 0; j < 2; ++j) {                 // B: lane holds column
      int col = wn * 32 + j * 16 + rlo;
      bfr[j].q[0] = *(const float4*)&Bs[cur][col][sel * 8];
      bfr[j].q[1] = *(const float4*)&Bs[cur][col][16 + sel * 8];
    }
#pragma unroll
    for (int i = 0; i < 4; ++i)
#pragma unroll
      for (int j = 0; j < 2; ++j)
        acc[i][j] = __builtin_amdgcn_wmma_f32_16x16x32_bf16(
            false, af[i].v, false, bfr[j].v, (short)0, acc[i][j], false, false);

    if (t + 1 < nk) {
#if USE_ASYNC_LDS
      __builtin_amdgcn_s_wait_asynccnt(0);
#endif
      __syncthreads();
    }
  }

#pragma unroll
  for (int i = 0; i < 4; ++i)
#pragma unroll
    for (int j = 0; j < 2; ++j)
#pragma unroll
      for (int r = 0; r < 8; ++r) {
        int m = m0 + wm * 64 + i * 16 + sel * 8 + r;
        int n = n0 + wn * 32 + j * 16 + rlo;
        float v = acc[i][j][r] * alpha;
        if (bias_m) v += bias_m[m];
        long idx = cBase + (long)m * ldc + n;
        if (Cf) Cf[idx] = v;
        if (Cb) Cb[idx] = (bf16)v;
      }
}

// ---------------------------------------------------------------------------
// Prep kernels
// ---------------------------------------------------------------------------
// zero the 2*6 pad rows of a [B][1036][256] bf16 buffer
__global__ void zero_pad(bf16* X) {
  int idx = blockIdx.x * blockDim.x + threadIdx.x;     // 32*12*256
  int b = idx / (12 * 256);
  int r = (idx / 256) % 12;
  int c = idx % 256;
  int row = (r < PADR) ? r : (LPAD - 12 + r);
  X[(long)b * LPAD * 256 + (long)row * 256 + c] = (bf16)0.0f;
}

// f32 x[b][c][l] -> bf16 Xt[b][6+l][c]
__global__ void transpose_pad(const float* __restrict__ x, bf16* __restrict__ Xt) {
  long idx = (long)blockIdx.x * blockDim.x + threadIdx.x;   // 32*256*1024
  int l = idx & 1023;
  int c = (idx >> 10) & 255;
  int b = (int)(idx >> 18);
  Xt[(long)b * LPAD * 256 + (long)(l + PADR) * 256 + c] = (bf16)x[idx];
}

// f32 w[m][c][13] -> bf16 W[m][dk*256 + c]   (implicit-GEMM conv weights)
__global__ void permute_convw(const float* __restrict__ w, bf16* __restrict__ W) {
  int idx = blockIdx.x * blockDim.x + threadIdx.x;     // 256*256*13
  int dk = idx % KS;
  int c  = (idx / KS) % DT;
  int m  = idx / (KS * DT);
  W[(long)m * KCONV + dk * DT + c] = (bf16)w[idx];
}

// f32 w[h][l][d] -> bf16 Wt[h][d][l]
__global__ void transpose_head(const float* __restrict__ w, bf16* __restrict__ Wt) {
  long idx = (long)blockIdx.x * blockDim.x + threadIdx.x;   // 8*1024*256
  int d = idx & 255;
  int l = (idx >> 8) & 1023;
  int h = (int)(idx >> 18);
  Wt[(long)h * DM * DT + (long)d * DM + l] = (bf16)w[idx];
}

__global__ void cvt_bf16(const float* __restrict__ s, bf16* __restrict__ d, int n) {
  int i = blockIdx.x * blockDim.x + threadIdx.x;
  if (i < n) d[i] = (bf16)s[i];
}

// ---------------------------------------------------------------------------
// Softmax: one wave32 per 256-wide row; normalizes f32 in place and emits bf16
// ---------------------------------------------------------------------------
__global__ __launch_bounds__(256)
void softmax_rows(float* __restrict__ S, bf16* __restrict__ Ab) {
  int wave = threadIdx.x >> 5, lane = threadIdx.x & 31;
  long row = (long)blockIdx.x * 8 + wave;
  float* r = S + row * 256;
  float v[8], mx = -3.4e38f;
#pragma unroll
  for (int i = 0; i < 8; ++i) { v[i] = r[lane + 32 * i]; mx = fmaxf(mx, v[i]); }
#pragma unroll
  for (int d = 16; d >= 1; d >>= 1) mx = fmaxf(mx, __shfl_xor(mx, d, 32));
  float s = 0.f;
#pragma unroll
  for (int i = 0; i < 8; ++i) { v[i] = __expf(v[i] - mx); s += v[i]; }
#pragma unroll
  for (int d = 16; d >= 1; d >>= 1) s += __shfl_xor(s, d, 32);
  float inv = 1.0f / s;
#pragma unroll
  for (int i = 0; i < 8; ++i) {
    float o = v[i] * inv;
    r[lane + 32 * i] = o;
    Ab[row * 256 + lane + 32 * i] = (bf16)o;
  }
}

// ---------------------------------------------------------------------------
// Residual + custom LayerNorm (ddof=1, eps added to sigma); block per row
// ---------------------------------------------------------------------------
__global__ __launch_bounds__(256)
void residual_ln(const float* __restrict__ F, const float* __restrict__ q,
                 const float* __restrict__ la, const float* __restrict__ lb,
                 float* __restrict__ out) {
  long base = (long)blockIdx.x * DM;
  int t = threadIdx.x, lane = t & 31, wave = t >> 5;
  float z[4], s = 0.f, sq = 0.f;
#pragma unroll
  for (int i = 0; i < 4; ++i) {
    float val = F[base + t + 256 * i] + q[base + t + 256 * i];
    z[i] = val; s += val; sq += val * val;
  }
#pragma unroll
  for (int d = 16; d >= 1; d >>= 1) { s += __shfl_xor(s, d, 32); sq += __shfl_xor(sq, d, 32); }
  __shared__ float ss[8], sg[8];
  if (lane == 0) { ss[wave] = s; sg[wave] = sq; }
  __syncthreads();
  s = 0.f; sq = 0.f;
#pragma unroll
  for (int w = 0; w < 8; ++w) { s += ss[w]; sq += sg[w]; }
  float mu  = s / (float)DM;
  float var = (sq - (float)DM * mu * mu) / (float)(DM - 1);
  float inv = 1.0f / (sqrtf(fmaxf(var, 0.f)) + 1e-3f);
#pragma unroll
  for (int i = 0; i < 4; ++i) {
    int l = t + 256 * i;
    out[base + l] = (z[i] - mu) * inv * la[l] + lb[l];
  }
}

// ---------------------------------------------------------------------------
extern "C" void kernel_launch(void* const* d_in, const int* in_sizes, int n_in,
                              void* d_out, int out_size, void* d_ws, size_t ws_size,
                              hipStream_t stream) {
  const float* q    = (const float*)d_in[0];
  const float* k    = (const float*)d_in[1];
  const float* v    = (const float*)d_in[2];
  const float* cq_w = (const float*)d_in[3];  const float* cq_b = (const float*)d_in[4];
  const float* ck_w = (const float*)d_in[5];  const float* ck_b = (const float*)d_in[6];
  const float* cv_w = (const float*)d_in[7];  const float* cv_b = (const float*)d_in[8];
  const float* w_qs = (const float*)d_in[9];
  const float* w_ks = (const float*)d_in[10];
  const float* w_vs = (const float*)d_in[11];
  const float* pw   = (const float*)d_in[12]; const float* pb   = (const float*)d_in[13];
  const float* fc_w = (const float*)d_in[14]; const float* fc_b = (const float*)d_in[15];
  const float* ln_a = (const float*)d_in[16]; const float* ln_b = (const float*)d_in[17];

  float* outF  = (float*)d_out;                     // (32,256,1024)
  float* attnF = outF + (long)B_ * DT * DM;         // (8*32,256,256)

  // workspace layout (bf16 unless noted)
  char* w8 = (char*)d_ws;
  size_t off = 0;
  auto take = [&](size_t bytes) { void* p = w8 + off; off = (off + bytes + 255) & ~(size_t)255; return p; };
  const size_t szXt   = (size_t)B_ * LPAD * DT * 2;
  const size_t szWc   = (size_t)DT * KCONV * 2;
  const size_t szWh   = (size_t)HEADS * DT * DM * 2;
  const size_t szXb   = (size_t)B_ * DT * DM * 2;
  const size_t szHS   = (size_t)HEADS * B_ * DT * DT * 2;

  bf16* xtq = (bf16*)take(szXt);
  bf16* xtk = (bf16*)take(szXt);
  bf16* xtv = (bf16*)take(szXt);
  bf16* wcq = (bf16*)take(szWc);
  bf16* wck = (bf16*)take(szWc);
  bf16* wcv = (bf16*)take(szWc);
  bf16* wfc = (bf16*)take(szWc);
  bf16* wqt = (bf16*)take(szWh);
  bf16* wkt = (bf16*)take(szWh);
  bf16* wvt = (bf16*)take(szWh);
  bf16* pwb = (bf16*)take((size_t)DM * HEADS * DT * 2);
  bf16* qb  = (bf16*)take(szXb);
  bf16* kb  = (bf16*)take(szXb);
  bf16* vb  = (bf16*)take(szXb);
  bf16* qs  = (bf16*)take(szHS);
  bf16* ks  = (bf16*)take(szHS);
  bf16* vst = (bf16*)take(szHS);
  bf16* atb = (bf16*)take(szHS);
  bf16* O   = (bf16*)take((size_t)B_ * DT * (HEADS * DT) * 2);
  bf16* ptp = (bf16*)take(szXt);
  float* F  = (float*)take((size_t)B_ * DT * DM * 4);
  (void)ws_size; (void)n_in; (void)in_sizes; (void)out_size;

  // ---- prep ----
  zero_pad<<<(B_ * 12 * 256) / 256, 256, 0, stream>>>(xtq);
  zero_pad<<<(B_ * 12 * 256) / 256, 256, 0, stream>>>(xtk);
  zero_pad<<<(B_ * 12 * 256) / 256, 256, 0, stream>>>(xtv);
  zero_pad<<<(B_ * 12 * 256) / 256, 256, 0, stream>>>(ptp);
  transpose_pad<<<(B_ * DT * DM) / 256, 256, 0, stream>>>(q, xtq);
  transpose_pad<<<(B_ * DT * DM) / 256, 256, 0, stream>>>(k, xtk);
  transpose_pad<<<(B_ * DT * DM) / 256, 256, 0, stream>>>(v, xtv);
  permute_convw<<<(DT * DT * KS) / 256, 256, 0, stream>>>(cq_w, wcq);
  permute_convw<<<(DT * DT * KS) / 256, 256, 0, stream>>>(ck_w, wck);
  permute_convw<<<(DT * DT * KS) / 256, 256, 0, stream>>>(cv_w, wcv);
  permute_convw<<<(DT * DT * KS) / 256, 256, 0, stream>>>(fc_w, wfc);
  transpose_head<<<(HEADS * DM * DT) / 256, 256, 0, stream>>>(w_qs, wqt);
  transpose_head<<<(HEADS * DM * DT) / 256, 256, 0, stream>>>(w_ks, wkt);
  transpose_head<<<(HEADS * DM * DT) / 256, 256, 0, stream>>>(w_vs, wvt);
  cvt_bf16<<<(DM * HEADS * DT) / 256, 256, 0, stream>>>(pw, pwb, DM * HEADS * DT);

  const long HSe = (long)DT * DT;           // per (h,b) 256x256 elems
  const long Xbe = (long)DT * DM;           // per b 256x1024
  // ---- S1: conv q/k/v as implicit GEMM, K = 13*256, ldb = 256 ----
  gemm_nt_bf16<<<dim3(8, 2, 32), 256, 0, stream>>>(wcq, xtq, nullptr, qb,
      DT, DM, KCONV, KCONV, DT, DM, 32, 0, 0, 0, (long)LPAD * DT, 0, Xbe, cq_b, 1.0f);
  gemm_nt_bf16<<<dim3(8, 2, 32), 256, 0, stream>>>(wck, xtk, nullptr, kb,
      DT, DM, KCONV, KCONV, DT, DM, 32, 0, 0, 0, (long)LPAD * DT, 0, Xbe, ck_b, 1.0f);
  gemm_nt_bf16<<<dim3(8, 2, 32), 256, 0, stream>>>(wcv, xtv, nullptr, vb,
      DT, DM, KCONV, KCONV, DT, DM, 32, 0, 0, 0, (long)LPAD * DT, 0, Xbe, cv_b, 1.0f);
  // ---- S2: per-head projections (z = h*32+b) ----
  gemm_nt_bf16<<<dim3(2, 2, 256), 256, 0, stream>>>(qb, wqt, nullptr, qs,
      DT, DT, DM, DM, DM, DT, 32, 0, Xbe, (long)DT * DM, 0, 32 * HSe, HSe, nullptr, 1.0f);
  gemm_nt_bf16<<<dim3(2, 2, 256), 256, 0, stream>>>(kb, wkt, nullptr, ks,
      DT, DT, DM, DM, DM, DT, 32, 0, Xbe, (long)DT * DM, 0, 32 * HSe, HSe, nullptr, 1.0f);
  gemm_nt_bf16<<<dim3(2, 2, 256), 256, 0, stream>>>(wvt, vb, nullptr, vst,   // v_s^T[d][k]
      DT, DT, DM, DM, DM, DT, 32, (long)DT * DM, 0, 0, Xbe, 32 * HSe, HSe, nullptr, 1.0f);
  // ---- S3: scores = q_s k_s^T / 32, f32 straight into d_out attn region ----
  gemm_nt_bf16<<<dim3(2, 2, 256), 256, 0, stream>>>(qs, ks, attnF, nullptr,
      DT, DT, DT, DT, DT, DT, 32, 32 * HSe, HSe, 32 * HSe, HSe, 32 * HSe, HSe,
      nullptr, 1.0f / 32.0f);
  // ---- S4: softmax rows (in place) + bf16 copy ----
  softmax_rows<<<(HEADS * B_ * DT) / 8, 256, 0, stream>>>(attnF, atb);
  // ---- S5: out = attn @ v_s, written as O[b][q][h*256+d] ----
  gemm_nt_bf16<<<dim3(2, 2, 256), 256, 0, stream>>>(atb, vst, nullptr, O,
      DT, DT, DT, DT, DT, HEADS * DT, 32, 32 * HSe, HSe, 32 * HSe, HSe,
      (long)DT, (long)DT * HEADS * DT, nullptr, 1.0f);
  // ---- S6: proj, transposed output into padded buffer for fc conv ----
  gemm_nt_bf16<<<dim3(2, 8, 32), 256, 0, stream>>>(pwb, O, nullptr, ptp + PADR * DT,
      DM, DT, HEADS * DT, HEADS * DT, HEADS * DT, DT, 32,
      0, 0, 0, (long)DT * HEADS * DT, 0, (long)LPAD * DT, pb, 1.0f);
  // ---- S7: fc conv as implicit GEMM ----
  gemm_nt_bf16<<<dim3(8, 2, 32), 256, 0, stream>>>(wfc, ptp, F, nullptr,
      DT, DM, KCONV, KCONV, DT, DM, 32, 0, 0, 0, (long)LPAD * DT, 0, Xbe, fc_b, 1.0f);
  // ---- S8: residual + custom LayerNorm ----
  residual_ln<<<B_ * DT, 256, 0, stream>>>(F, q, ln_a, ln_b, outF);
}
